// ADDer_55937654063700
// MI455X (gfx1250) — matrix operand order repaired and checked by reading, not compiled, not run on hardware
//
#include <hip/hip_runtime.h>

#define F 128
#define BLK_ROWS 64

typedef __attribute__((ext_vector_type(16))) __bf16 v16bf;
typedef __attribute__((ext_vector_type(8)))  float  v8f;
typedef __attribute__((ext_vector_type(16))) unsigned short v16u;
typedef __attribute__((ext_vector_type(8)))  unsigned short v8u;

// round-to-nearest-even fp32 -> bf16 (bit-level)
static __device__ __forceinline__ unsigned short f2bf(float f) {
    unsigned u = __builtin_bit_cast(unsigned, f);
    unsigned r = u + 0x7FFFu + ((u >> 16) & 1u);
    return (unsigned short)(r >> 16);
}

static __device__ __forceinline__ v8u pack8(float4 p0, float4 p1, float s) {
    v8u pk;
    pk[0] = f2bf(p0.x * s); pk[1] = f2bf(p0.y * s);
    pk[2] = f2bf(p0.z * s); pk[3] = f2bf(p0.w * s);
    pk[4] = f2bf(p1.x * s); pk[5] = f2bf(p1.y * s);
    pk[6] = f2bf(p1.z * s); pk[7] = f2bf(p1.w * s);
    return pk;
}

__global__ __launch_bounds__(256) void cvt_bf16_kernel(const float* __restrict__ in,
                                                       unsigned short* __restrict__ out, int n) {
    int i = blockIdx.x * blockDim.x + threadIdx.x;
    if (i < n) out[i] = f2bf(in[i]);
}

__global__ __launch_bounds__(256) void degree_kernel(const long long* __restrict__ dst,
                                                     float* __restrict__ deg, int n_edges) {
    int e = blockIdx.x * blockDim.x + threadIdx.x;
    if (e < n_edges) atomicAdd(&deg[(int)dst[e]], 1.0f);
}

// one block = 2 edges, 128 lanes per edge (one feature each)
__global__ __launch_bounds__(256) void scatter_add_kernel(const float* __restrict__ feat,
                                                          const long long* __restrict__ src,
                                                          const long long* __restrict__ dst,
                                                          float* __restrict__ sum, int n_edges) {
    int e = blockIdx.x * 2 + (threadIdx.x >> 7);
    int f = threadIdx.x & 127;
    if (e < n_edges) {
        int s = (int)src[e];
        int d = (int)dst[e];
        atomicAdd(&sum[(size_t)d * F + f], feat[(size_t)s * F + f]);
    }
}

// out = (A*scale) @ Wl^T + bias (+ X @ Wr^T) (+ReLU), scale = 1/max(deg,1) if HAS_DEG.
// Block: 64 rows x 128 cols, 8 waves. Each wave owns a 16-col tile, preloads all
// B fragments (Wl/Wr x 4 K-chunks) into VGPRs once, then sweeps 4 M-subtiles of
// 16 rows each -> 32 straight-line WMMAs per wave, A fragments from LDS.
template<bool HAS_X, bool HAS_DEG, bool RELU>
__global__ __launch_bounds__(256) void sage_gemm_kernel(
    const float* __restrict__ A,
    const float* __restrict__ deg,
    const unsigned short* __restrict__ Wl,   // bf16 [F][F], row n holds W[n][k]
    const float* __restrict__ bias,          // [F]
    const float* __restrict__ X,             // [N,F]
    const unsigned short* __restrict__ Wr,   // bf16 [F][F]
    float* __restrict__ Out,
    int n_nodes)
{
    __shared__ unsigned short sA[BLK_ROWS][F];
    __shared__ unsigned short sX[HAS_X ? BLK_ROWS : 1][F];

    const int row0 = blockIdx.x * BLK_ROWS;
    const int tid  = threadIdx.x;
    // uniform: every block except the last is full -> straight-line store path
    const bool full = (row0 + BLK_ROWS) <= n_nodes;

    // ---- stage 64x128 A (and X) tile into LDS as bf16; clamp rows past N so
    //      EXEC stays all-ones for the WMMA region ----
    {
        const int r  = tid >> 2;             // 0..63
        const int c0 = (tid & 3) * 32;       // 32 features per thread
        int node = row0 + r;
        if (node >= n_nodes) node = n_nodes - 1;
        float scale = 1.0f;
        if (HAS_DEG) scale = 1.0f / fmaxf(deg[node], 1.0f);
        const float4* a4 = (const float4*)(A + (size_t)node * F + c0);
        #pragma unroll
        for (int j = 0; j < 4; ++j)
            *(v8u*)&sA[r][c0 + 8 * j] = pack8(a4[2 * j], a4[2 * j + 1], scale);
        if (HAS_X) {
            const float4* x4 = (const float4*)(X + (size_t)node * F + c0);
            #pragma unroll
            for (int j = 0; j < 4; ++j)
                *(v8u*)&sX[r][c0 + 8 * j] = pack8(x4[2 * j], x4[2 * j + 1], 1.0f);
        }
    }
    __syncthreads();

    const int wave = tid >> 5;               // 0..7 -> 16-col tile
    const int lane = tid & 31;
    const int kb   = (lane & 16) ? 8 : 0;    // A frag K base (ISA 16-bit A layout)
    const int kh   = (lane & 16) ? 8 : 0;    // B frag K base (same K order as A frag)
    const int ncol = wave * 16 + (lane & 15);

    // ---- preload all B fragments into registers (reused across 4 M-subtiles) ----
    v16u bl[4], br[4];
    #pragma unroll
    for (int kc = 0; kc < 4; ++kc) {
        const v8u* wl = (const v8u*)(Wl + (size_t)ncol * F + kc * 32 + kh);
        v8u b0 = wl[0], b1 = wl[2];
        #pragma unroll
        for (int i = 0; i < 8; ++i) { bl[kc][i] = b0[i]; bl[kc][i + 8] = b1[i]; }
        if (HAS_X) {
            const v8u* wr = (const v8u*)(Wr + (size_t)ncol * F + kc * 32 + kh);
            v8u c0v = wr[0], c1v = wr[2];
            #pragma unroll
            for (int i = 0; i < 8; ++i) { br[kc][i] = c0v[i]; br[kc][i + 8] = c1v[i]; }
        }
    }

    const float bv   = bias[ncol];
    const int   mofs = (lane & 16) ? 8 : 0;

    #pragma unroll
    for (int sub = 0; sub < 4; ++sub) {
        const int mrow = sub * 16 + (lane & 15);
        v8f acc = {};
        #pragma unroll
        for (int kc = 0; kc < 4; ++kc) {
            v8u alo = *(const v8u*)&sA[mrow][kc * 32 + kb];
            v8u ahi = *(const v8u*)&sA[mrow][kc * 32 + kb + 16];
            v16u au;
            #pragma unroll
            for (int i = 0; i < 8; ++i) { au[i] = alo[i]; au[i + 8] = ahi[i]; }
            acc = __builtin_amdgcn_wmma_f32_16x16x32_bf16(
                false, __builtin_bit_cast(v16bf, au),
                false, __builtin_bit_cast(v16bf, bl[kc]),
                (short)0, acc, false, false);
            if (HAS_X) {
                v8u xlo = *(const v8u*)&sX[mrow][kc * 32 + kb];
                v8u xhi = *(const v8u*)&sX[mrow][kc * 32 + kb + 16];
                v16u xu;
                #pragma unroll
                for (int i = 0; i < 8; ++i) { xu[i] = xlo[i]; xu[i + 8] = xhi[i]; }
                acc = __builtin_amdgcn_wmma_f32_16x16x32_bf16(
                    false, __builtin_bit_cast(v16bf, xu),
                    false, __builtin_bit_cast(v16bf, br[kc]),
                    (short)0, acc, false, false);
            }
        }
        // epilogue: bias, ReLU, store (C/D layout: lane->N, VGPR->M).
        // Uniform fast path: full blocks store straight-line (no exec churn).
        float* outp = Out + (size_t)(row0 + sub * 16 + mofs) * F + ncol;
        if (full) {
            #pragma unroll
            for (int v = 0; v < 8; ++v) {
                float val = acc[v] + bv;
                if (RELU) val = fmaxf(val, 0.0f);
                outp[(size_t)v * F] = val;
            }
        } else {
            #pragma unroll
            for (int v = 0; v < 8; ++v) {
                int row = row0 + sub * 16 + v + mofs;
                float val = acc[v] + bv;
                if (RELU) val = fmaxf(val, 0.0f);
                if (row < n_nodes) outp[(size_t)v * F] = val;
            }
        }
    }
}

extern "C" void kernel_launch(void* const* d_in, const int* in_sizes, int n_in,
                              void* d_out, int out_size, void* d_ws, size_t ws_size,
                              hipStream_t stream) {
    const float*      x     = (const float*)d_in[0];
    const long long*  xedge = (const long long*)d_in[1];   // int64 [2,E]
    const float*      w1_l  = (const float*)d_in[2];
    const float*      b1_l  = (const float*)d_in[3];
    const float*      w1_r  = (const float*)d_in[4];
    const float*      w2_l  = (const float*)d_in[5];
    const float*      b2_l  = (const float*)d_in[6];
    const float*      w2_r  = (const float*)d_in[7];
    const float*      w_dec = (const float*)d_in[8];
    const float*      b_dec = (const float*)d_in[9];

    const int N = in_sizes[0] / F;     // 100000
    const int E = in_sizes[1] / 2;     // 600000
    const long long* srcE = xedge;
    const long long* dstE = xedge + E;

    // workspace layout
    float* ws  = (float*)d_ws;
    float* deg = ws;                                   // [N]
    float* sum = ws + (size_t)256 * ((N + 255) / 256); // [N*F]
    float* h1  = sum + (size_t)N * F;                  // [N*F]
    unsigned short* wb   = (unsigned short*)(h1 + (size_t)N * F);
    unsigned short* w1lb = wb;
    unsigned short* w1rb = wb + 1 * (F * F);
    unsigned short* w2lb = wb + 2 * (F * F);
    unsigned short* w2rb = wb + 3 * (F * F);
    unsigned short* wdcb = wb + 4 * (F * F);

    float* h  = (float*)d_out;             // encoder output [N*F]
    float* dx = h + (size_t)N * F;         // decoder output [N*F]

    const int gemm_grid = (N + BLK_ROWS - 1) / BLK_ROWS;

    // pre-convert weights to bf16 (tiny; stays L2-resident)
    cvt_bf16_kernel<<<(F * F) / 256, 256, 0, stream>>>(w1_l,  w1lb, F * F);
    cvt_bf16_kernel<<<(F * F) / 256, 256, 0, stream>>>(w1_r,  w1rb, F * F);
    cvt_bf16_kernel<<<(F * F) / 256, 256, 0, stream>>>(w2_l,  w2lb, F * F);
    cvt_bf16_kernel<<<(F * F) / 256, 256, 0, stream>>>(w2_r,  w2rb, F * F);
    cvt_bf16_kernel<<<(F * F) / 256, 256, 0, stream>>>(w_dec, wdcb, F * F);

    // degrees (shared by both layers)
    hipMemsetAsync(deg, 0, sizeof(float) * (size_t)N, stream);
    degree_kernel<<<(E + 255) / 256, 256, 0, stream>>>(dstE, deg, E);

    // ---- layer 1: h1 = relu(mean_agg(x) @ w1_l^T + b1_l + x @ w1_r^T) ----
    hipMemsetAsync(sum, 0, sizeof(float) * (size_t)N * F, stream);
    scatter_add_kernel<<<(E + 1) / 2, 256, 0, stream>>>(x, srcE, dstE, sum, E);
    sage_gemm_kernel<true, true, true><<<gemm_grid, 256, 0, stream>>>(
        sum, deg, w1lb, b1_l, x, w1rb, h1, N);

    // ---- layer 2: h = mean_agg(h1) @ w2_l^T + b2_l + h1 @ w2_r^T ----
    hipMemsetAsync(sum, 0, sizeof(float) * (size_t)N * F, stream);
    scatter_add_kernel<<<(E + 1) / 2, 256, 0, stream>>>(h1, srcE, dstE, sum, E);
    sage_gemm_kernel<true, true, false><<<gemm_grid, 256, 0, stream>>>(
        sum, deg, w2lb, b2_l, h1, w2rb, h, N);

    // ---- decoder: dx = h @ w_dec^T + b_dec ----
    sage_gemm_kernel<false, false, false><<<gemm_grid, 256, 0, stream>>>(
        h, nullptr, wdcb, b_dec, nullptr, nullptr, dx, N);
}
